// HieraWeatherEmbedding_66864050864882
// MI455X (gfx1250) — matrix operand order repaired and checked by reading, not compiled, not run on hardware
//
#include <hip/hip_runtime.h>

// ---------------- problem constants ----------------
#define L_TOT   8192
#define NV      88
#define DDIM    128
#define NGRP    12
#define EDIM    1536
#define SCALE_A 0.125f   // 64^-0.5
#define LTILE   16

typedef __attribute__((ext_vector_type(16))) __bf16 v16bf;
typedef __attribute__((ext_vector_type(8)))  float  v8f;
typedef __attribute__((ext_vector_type(4)))  unsigned int v4u;
typedef __attribute__((ext_vector_type(8)))  int   v8i;
typedef __attribute__((ext_vector_type(4)))  int   v4i;

__constant__ int g_len[NGRP] = {7,7,7,7,7,9,9,9,9,9,17,6};
__constant__ int g_ids[NGRP][17] = {
  {3,8,13,18,23,28,33,0,0,0,0,0,0,0,0,0,0},
  {4,9,14,19,24,29,34,0,0,0,0,0,0,0,0,0,0},
  {2,7,12,17,22,27,32,0,0,0,0,0,0,0,0,0,0},
  {1,6,11,16,21,26,31,0,0,0,0,0,0,0,0,0,0},
  {0,5,10,15,20,25,30,0,0,0,0,0,0,0,0,0,0},
  {3,8,13,18,23,28,33,38,43,0,0,0,0,0,0,0,0},
  {4,9,14,19,24,29,34,39,44,0,0,0,0,0,0,0,0},
  {2,7,12,17,22,27,32,37,42,0,0,0,0,0,0,0,0},
  {1,6,11,16,21,26,31,36,41,0,0,0,0,0,0,0,0},
  {0,5,10,15,20,25,30,35,40,0,0,0,0,0,0,0,0},
  {65,66,67,68,69,70,71,72,73,74,75,76,77,78,79,80,81},
  {82,83,84,85,86,87,0,0,0,0,0,0,0,0,0,0,0}
};

// ---------------- workspace layout (bytes) ----------------
static const size_t XN_OFF  = 0;            // bf16 [8192][88][128]  184,549,376 B
static const size_t OUTG_OFF= 184549376;    // bf16 [8192][12][128]   25,165,824 B
static const size_t QG_OFF  = 209715200;    // f32  [12][128]
static const size_t QE_OFF  = 209721344;    // f32  [12][128]
static const size_t QKV_OFF = 209727488;    // f32  [12][2][128]
static const size_t QC_OFF  = 209739776;    // f32  [12][2]
static const size_t QK2_OFF = 209740032;    // f32  [12][2][128]
static const size_t CQ2_OFF = 209752320;    // f32  [12][2]
static const size_t ABF_OFF = 209752576;    // bf16 [12][2][128][128]
static const size_t EBF_OFF = 210539008;    // bf16 [2][128][128]
static const size_t YB_OFF  = 210604544;    // f32  [12][128]
static const size_t RB_OFF  = 210610688;    // f32  [128]

// ---------------- WMMA fragment helpers (per ISA 7.12.2 layouts) ----------------
// A (16x32, 16-bit): lanes 0-15 M=lane, lanes 16-31 M=lane-16 with K shifted +8.
__device__ __forceinline__ v16bf load_fragA_b16(const __bf16* row, int kbase, int lane) {
  const int k0 = kbase + ((lane & 16) ? 8 : 0);
  v16bf f;
#pragma unroll
  for (int j = 0; j < 8; ++j) f[j] = row[k0 + j];
#pragma unroll
  for (int j = 0; j < 8; ++j) f[8 + j] = row[k0 + 16 + j];
  return f;
}
// B (32x16, 16-bit): N = lane&15; same K order as A (lanes 16-31 K shifted +8).
// 'col' points at storage[n][k] row (row-major over k).
__device__ __forceinline__ v16bf load_fragB_b16(const __bf16* col, int kbase, int lane) {
  const int k0 = kbase + ((lane & 16) ? 8 : 0);
  v16bf f;
#pragma unroll
  for (int j = 0; j < 8; ++j) f[j] = col[k0 + j];
#pragma unroll
  for (int j = 0; j < 8; ++j) f[8 + j] = col[k0 + 16 + j];
  return f;
}

// ---------------- TDM (Tensor Data Mover) contiguous copy, ISA ch.8 D# ----------------
#if __has_builtin(__builtin_amdgcn_tensor_load_to_lds)
#define USE_TDM 1
#else
#define USE_TDM 0
#endif

#if USE_TDM
__device__ __forceinline__ void tdm_copy_contig(unsigned lds_byte_off, const void* src,
                                                unsigned units8 /* 8-byte units, <=65535 */) {
  unsigned long long ga = (unsigned long long)src;
  // group0: count=1, lds_addr, global_addr[56:0], type=2
  v4u g0 = { 1u, lds_byte_off,
             (unsigned)(ga & 0xffffffffu),
             (unsigned)((ga >> 32) | 0x80000000u) };
  // group1: data_size=8B; tensor_dim0=units8; tensor_dim1=1; tile_dim0=units8;
  //         tile_dim1/2=0 (unused); tensor_dim0_stride=units8
  v8i g1 = { (int)(3u << 16),
             (int)((units8 & 0xffffu) << 16),
             (int)((units8 >> 16) | (1u << 16)),
             (int)((units8 & 0xffffu) << 16),
             0,
             (int)units8,
             0, 0 };
  v4i g2 = { 0, 0, 0, 0 };
  v4i g3 = { 0, 0, 0, 0 };
#if defined(__clang_major__) && (__clang_major__ >= 23)
  v8i gz = { 0,0,0,0,0,0,0,0 };
  __builtin_amdgcn_tensor_load_to_lds(g0, g1, g2, g3, gz, 0);
#else
  __builtin_amdgcn_tensor_load_to_lds(g0, g1, g2, g3, 0);
#endif
#if __has_builtin(__builtin_amdgcn_s_wait_tensorcnt)
  __builtin_amdgcn_s_wait_tensorcnt(0);
#else
  asm volatile("s_wait_tensorcnt 0" ::: "memory");
#endif
}
#endif

// ---------------- kernel 1: patch embed -> xn bf16 [l][v][128] ----------------
__global__ __launch_bounds__(256) void k_embed(const float* __restrict__ x,
                                               const float* __restrict__ conv_w,
                                               const float* __restrict__ conv_b,
                                               const float* __restrict__ cemb,
                                               __bf16* __restrict__ xn)
{
  size_t idx = (size_t)blockIdx.x * 256 + threadIdx.x;
  if (idx >= (size_t)L_TOT * NV * DDIM) return;
  int d = (int)(idx & 127);
  int v = (int)((idx >> 7) % NV);
  int l = (int)(idx / (128 * NV));
  int hi = l >> 7;        // l / 128 (wp = 128)
  int wj = l & 127;
  const float* xb = x + ((size_t)v * 128 + 2 * hi) * 256 + 2 * wj;
  float p0 = xb[0], p1 = xb[1], p2 = xb[256], p3 = xb[257];
  const float* w = conv_w + ((size_t)v * 4) * 128 + d;
  float acc = p0 * w[0] + p1 * w[128] + p2 * w[256] + p3 * w[384];
  acc += conv_b[v * 128 + d] + cemb[v * 128 + d];
  xn[((size_t)l * NV + v) * 128 + d] = (__bf16)acc;
}

// ---------------- precompute kernels ----------------
__global__ __launch_bounds__(256) void k_pre_q(const float* __restrict__ cquery,
                                               const float* __restrict__ grp_in_w,
                                               const float* __restrict__ grp_in_b,
                                               const float* __restrict__ equery,
                                               const float* __restrict__ ens_in_w,
                                               const float* __restrict__ ens_in_b,
                                               float* __restrict__ qg, float* __restrict__ qe)
{
  int idx = blockIdx.x * 256 + threadIdx.x;
  if (idx >= 2 * NGRP * 128) return;
  int d = idx & 127;
  int i = (idx >> 7) % NGRP;
  if (idx < NGRP * 128) {
    const float* Wq = grp_in_w + ((size_t)i * 384 + d) * 128;
    const float* cq = cquery + i * 128;
    float a = 0.f;
    for (int e = 0; e < 128; ++e) a += cq[e] * Wq[e];
    qg[i * 128 + d] = a + grp_in_b[i * 384 + d];
  } else {
    const float* Wq = ens_in_w + (size_t)d * 128;
    const float* cq = equery + i * 128;
    float a = 0.f;
    for (int e = 0; e < 128; ++e) a += cq[e] * Wq[e];
    qe[i * 128 + d] = a + ens_in_b[d];
  }
}

__global__ __launch_bounds__(256) void k_pre_qk(const float* __restrict__ grp_in_w,
                                                const float* __restrict__ grp_in_b,
                                                const float* __restrict__ ens_in_w,
                                                const float* __restrict__ ens_in_b,
                                                const float* __restrict__ qg,
                                                const float* __restrict__ qe,
                                                float* __restrict__ qkvec, float* __restrict__ qconst,
                                                float* __restrict__ Qk2,   float* __restrict__ cq2)
{
  int idx = blockIdx.x * 256 + threadIdx.x;
  if (idx < 3072) {
    int c = idx & 127, h = (idx >> 7) & 1, i = idx >> 8;
    float a = 0.f;
    for (int d2 = 0; d2 < 64; ++d2)
      a += grp_in_w[((size_t)i * 384 + 128 + h * 64 + d2) * 128 + c] * qg[i * 128 + h * 64 + d2];
    qkvec[(i * 2 + h) * 128 + c] = a;
  } else if (idx < 6144) {
    int j = idx - 3072; int c = j & 127, h = (j >> 7) & 1, t = j >> 8;
    float a = 0.f;
    for (int d2 = 0; d2 < 64; ++d2)
      a += ens_in_w[((size_t)(128 + h * 64 + d2)) * 128 + c] * qe[t * 128 + h * 64 + d2];
    Qk2[(t * 2 + h) * 128 + c] = a;
  } else if (idx < 6168) {
    int j = idx - 6144; int h = j & 1, i = j >> 1;
    float a = 0.f;
    for (int d2 = 0; d2 < 64; ++d2)
      a += qg[i * 128 + h * 64 + d2] * grp_in_b[i * 384 + 128 + h * 64 + d2];
    qconst[i * 2 + h] = a;
  } else if (idx < 6192) {
    int j = idx - 6168; int h = j & 1, t = j >> 1;
    float a = 0.f;
    for (int d2 = 0; d2 < 64; ++d2)
      a += qe[t * 128 + h * 64 + d2] * ens_in_b[128 + h * 64 + d2];
    cq2[t * 2 + h] = a;
  }
}

__global__ __launch_bounds__(256) void k_pre_mats(const float* __restrict__ grp_in_w,
                                                  const float* __restrict__ grp_in_b,
                                                  const float* __restrict__ grp_out_w,
                                                  const float* __restrict__ grp_out_b,
                                                  const float* __restrict__ ens_in_w,
                                                  const float* __restrict__ ens_in_b,
                                                  const float* __restrict__ ens_out_w,
                                                  const float* __restrict__ ens_out_b,
                                                  __bf16* __restrict__ Abf, __bf16* __restrict__ Ebf,
                                                  float* __restrict__ ybias, float* __restrict__ rbias)
{
  int idx = blockIdx.x * 256 + threadIdx.x;
  if (idx < 393216) {                       // Abf[i][h][o][c] = grp_out_w[i][o][h64+d2]*Wv[h64+d2][c]
    int c = idx & 127, o = (idx >> 7) & 127, h = (idx >> 14) & 1, i = idx >> 15;
    float a = 0.f;
    for (int d2 = 0; d2 < 64; ++d2)
      a += grp_out_w[((size_t)i * 128 + o) * 128 + h * 64 + d2]
         * grp_in_w[((size_t)i * 384 + 256 + h * 64 + d2) * 128 + c];
    Abf[(((size_t)i * 2 + h) * 128 + o) * 128 + c] = (__bf16)a;
  } else if (idx < 425984) {                // Ebf[h][o][c]
    int j = idx - 393216; int c = j & 127, o = (j >> 7) & 127, h = j >> 14;
    float a = 0.f;
    for (int d2 = 0; d2 < 64; ++d2)
      a += ens_out_w[(size_t)o * 128 + h * 64 + d2]
         * ens_in_w[((size_t)(256 + h * 64 + d2)) * 128 + c];
    Ebf[((size_t)h * 128 + o) * 128 + c] = (__bf16)a;
  } else if (idx < 427520) {                // ybias[i][o] = grp_out_w@bv + grp_out_b
    int j = idx - 425984; int o = j & 127, i = j >> 7;
    float a = grp_out_b[i * 128 + o];
    for (int d = 0; d < 128; ++d)
      a += grp_out_w[((size_t)i * 128 + o) * 128 + d] * grp_in_b[i * 384 + 256 + d];
    ybias[i * 128 + o] = a;
  } else if (idx < 427648) {                // rbias[o]
    int o = idx - 427520;
    float a = ens_out_b[o];
    for (int d = 0; d < 128; ++d)
      a += ens_out_w[(size_t)o * 128 + d] * ens_in_b[256 + d];
    rbias[o] = a;
  }
}

// ---------------- kernel: per-group fused attention (1 wave / 16-l tile / group) ----------------
__global__ __launch_bounds__(32) void k_group_attn(const __bf16* __restrict__ xn,
                                                   const float* __restrict__ qkvec,
                                                   const float* __restrict__ qconst,
                                                   const __bf16* __restrict__ Abf,
                                                   const float* __restrict__ ybias,
                                                   __bf16* __restrict__ outg)
{
  extern __shared__ char smem[];
  const int grp = blockIdx.y;
  const int l0 = blockIdx.x * LTILE;
  const int lane = (int)threadIdx.x;
  const int ng = g_len[grp];

  __bf16* s_kv  = (__bf16*)(smem);                        // [16][ng][128]   (max 69632 B)
  float*  s_w   = (float*)(smem + 69632);                 // [16][2][17]     2176 B
  float*  s_qk  = (float*)(smem + 69632 + 2176);          // [2][128]        1024 B
  float*  s_qc  = (float*)(smem + 69632 + 2176 + 1024);   // [2]             16 B
  __bf16* s_ctx = (__bf16*)(smem + 69632 + 2176 + 1024 + 16); // [16][2][128] 8192 B

  for (int t = lane; t < 256; t += 32) s_qk[t] = qkvec[(grp * 2) * 128 + t];
  if (lane < 2) s_qc[lane] = qconst[grp * 2 + lane];

  // stage kv tile to LDS (rows of 128 bf16 = 16 x 16B chunks)
  const int nchunk = LTILE * ng * 16;
  for (int ch = lane; ch < nchunk; ch += 32) {
    int l   = ch / (ng * 16);
    int rem = ch - l * ng * 16;
    int g   = rem >> 4;
    int off = rem & 15;
    const uint4* src = (const uint4*)(xn + ((size_t)(l0 + l) * NV + g_ids[grp][g]) * 128) + off;
    uint4* dst = (uint4*)(s_kv + (size_t)(l * ng + g) * 128) + off;
    *dst = *src;
  }
  __syncthreads();

  // logits + softmax over group members; lane -> (l, h)
  {
    const int l = lane & 15, h = lane >> 4;
    const float* qk = s_qk + h * 128;
    float* wr = s_w + (l * 2 + h) * 17;
    for (int g = 0; g < ng; ++g) {
      const __bf16* kr = s_kv + (l * ng + g) * 128;
      float a = 0.f;
      for (int c = 0; c < 128; ++c) a += (float)kr[c] * qk[c];
      wr[g] = (a + s_qc[h]) * SCALE_A;
    }
    float mx = -3.0e38f;
    for (int g = 0; g < ng; ++g) mx = fmaxf(mx, wr[g]);
    float sum = 0.f;
    for (int g = 0; g < ng; ++g) { float e = __expf(wr[g] - mx); wr[g] = e; sum += e; }
    float inv = 1.f / sum;
    for (int g = 0; g < ng; ++g) wr[g] *= inv;
  }
  __syncthreads();

  // ctx[l][h][c] = sum_g w * kv
  for (int u = lane; u < 16 * 2 * 128; u += 32) {
    int c = u & 127, h = (u >> 7) & 1, l = u >> 8;
    float a = 0.f;
    for (int g = 0; g < ng; ++g)
      a += s_w[(l * 2 + h) * 17 + g] * (float)s_kv[(l * ng + g) * 128 + c];
    s_ctx[(l * 2 + h) * 128 + c] = (__bf16)a;
  }
  __syncthreads();

  // y = A0*ctx0 + A1*ctx1 + ybias via bf16 WMMA, M=16 l's, N=128 in 8 tiles, K=128 in 4 tiles
  v16bf af[2][4];
  {
    int m = lane & 15;
#pragma unroll
    for (int h = 0; h < 2; ++h)
#pragma unroll
      for (int kt = 0; kt < 4; ++kt)
        af[h][kt] = load_fragA_b16(s_ctx + (m * 2 + h) * 128, kt * 32, lane);
  }
  const int N = lane & 15;
  const int mB = (lane & 16) ? 8 : 0;
  const __bf16* Ag = Abf + (size_t)grp * 2 * 128 * 128;
#pragma unroll
  for (int nt = 0; nt < 8; ++nt) {
    float b0 = ybias[grp * 128 + nt * 16 + N];
    v8f acc;
#pragma unroll
    for (int r = 0; r < 8; ++r) acc[r] = b0;
#pragma unroll
    for (int h = 0; h < 2; ++h)
#pragma unroll
      for (int kt = 0; kt < 4; ++kt) {
        v16bf bf = load_fragB_b16(Ag + ((size_t)h * 128 + nt * 16 + N) * 128, kt * 32, lane);
        acc = __builtin_amdgcn_wmma_f32_16x16x32_bf16(false, af[h][kt], false, bf,
                                                      (short)0, acc, false, false);
      }
#pragma unroll
    for (int r = 0; r < 8; ++r) {
      int l = l0 + mB + r;
      outg[((size_t)l * NGRP + grp) * 128 + nt * 16 + N] = (__bf16)acc[r];
    }
  }
}

// ---------------- kernel: ensemble attention (1 wave / 16-l tile) ----------------
__global__ __launch_bounds__(32) void k_ens_attn(const __bf16* __restrict__ outg,
                                                 const float* __restrict__ Qk2,
                                                 const float* __restrict__ cq2,
                                                 const __bf16* __restrict__ Ebf,
                                                 const float* __restrict__ rbias,
                                                 float* __restrict__ r_out)
{
  extern __shared__ char smem[];
  const int l0 = blockIdx.x * LTILE;
  const int lane = (int)threadIdx.x;

  __bf16* s_out = (__bf16*)(smem);            // [16][12][128]  49152 B
  float*  s_qk  = (float*)(smem + 49152);     // [12][2][128]   12288 B
  float*  s_cq  = (float*)(smem + 61440);     // [24]           96 B
  float*  s_rb  = (float*)(smem + 61536);     // [128]          512 B
  float*  s_w   = (float*)(smem + 62048);     // [16][2][12][12]18432 B
  __bf16* s_ctx = (__bf16*)(smem + 80480);    // [16][2][128]   8192 B

  for (int t = lane; t < 12 * 2 * 128; t += 32) s_qk[t] = Qk2[t];
  for (int t = lane; t < 24; t += 32) s_cq[t] = cq2[t];
  for (int t = lane; t < 128; t += 32) s_rb[t] = rbias[t];

  // stage 'out' tile (contiguous 49152 B) -> LDS
  const __bf16* src = outg + (size_t)l0 * NGRP * 128;
#if USE_TDM
  tdm_copy_contig(__builtin_amdgcn_groupstaticsize(), src, 49152u / 8u);
#else
  for (int ch = lane; ch < 3072; ch += 32)
    ((uint4*)s_out)[ch] = ((const uint4*)src)[ch];
#endif
  __syncthreads();

  // logits + softmax over s (12); lane -> (l, h)
  const int l = lane & 15, h = lane >> 4;
  for (int t = 0; t < 12; ++t) {
    const float* qk = s_qk + (t * 2 + h) * 128;
    float* wr = s_w + ((l * 2 + h) * 12 + t) * 12;
    for (int s = 0; s < 12; ++s) {
      const __bf16* orow = s_out + (l * 12 + s) * 128;
      float a = 0.f;
      for (int c = 0; c < 128; ++c) a += (float)orow[c] * qk[c];
      wr[s] = (a + s_cq[t * 2 + h]) * SCALE_A;
    }
    float mx = -3.0e38f;
    for (int s = 0; s < 12; ++s) mx = fmaxf(mx, wr[s]);
    float sum = 0.f;
    for (int s = 0; s < 12; ++s) { float e = __expf(wr[s] - mx); wr[s] = e; sum += e; }
    float inv = 1.f / sum;
    for (int s = 0; s < 12; ++s) wr[s] *= inv;
  }
  __syncthreads();

  const int N = lane & 15;
  const int mB = (lane & 16) ? 8 : 0;
  for (int t = 0; t < 12; ++t) {
    for (int u = lane; u < 16 * 2 * 128; u += 32) {
      int c = u & 127, h2 = (u >> 7) & 1, l2 = u >> 8;
      const float* wr = s_w + ((l2 * 2 + h2) * 12 + t) * 12;
      float a = 0.f;
      for (int s = 0; s < 12; ++s) a += wr[s] * (float)s_out[(l2 * 12 + s) * 128 + c];
      s_ctx[(l2 * 2 + h2) * 128 + c] = (__bf16)a;
    }
    __syncthreads();

    v16bf af[2][4];
    {
      int m = lane & 15;
#pragma unroll
      for (int hh = 0; hh < 2; ++hh)
#pragma unroll
        for (int kt = 0; kt < 4; ++kt)
          af[hh][kt] = load_fragA_b16(s_ctx + (m * 2 + hh) * 128, kt * 32, lane);
    }
#pragma unroll
    for (int nt = 0; nt < 8; ++nt) {
      float b0 = s_rb[nt * 16 + N];
      v8f acc;
#pragma unroll
      for (int r = 0; r < 8; ++r) acc[r] = b0;
#pragma unroll
      for (int hh = 0; hh < 2; ++hh)
#pragma unroll
        for (int kt = 0; kt < 4; ++kt) {
          v16bf bf = load_fragB_b16(Ebf + ((size_t)hh * 128 + nt * 16 + N) * 128, kt * 32, lane);
          acc = __builtin_amdgcn_wmma_f32_16x16x32_bf16(false, af[hh][kt], false, bf,
                                                        (short)0, acc, false, false);
        }
#pragma unroll
      for (int r = 0; r < 8; ++r) {
        int ll = l0 + mB + r;
        r_out[((size_t)ll * NGRP + t) * 128 + nt * 16 + N] = acc[r];
      }
    }
    __syncthreads();
  }
}

// ---------------- kernel: LayerNorm over 1536, in place in d_out ----------------
__global__ __launch_bounds__(256) void k_layernorm(float* __restrict__ r_out,
                                                   const float* __restrict__ gamma,
                                                   const float* __restrict__ beta)
{
  __shared__ float s_sum[256];
  __shared__ float s_sq[256];
  const int lrow = blockIdx.x;
  const int tid = (int)threadIdx.x;
  float* row = r_out + (size_t)lrow * EDIM;
  float lsum = 0.f, lsq = 0.f;
  float vals[6];
#pragma unroll
  for (int j = 0; j < 6; ++j) {
    float v = row[tid + j * 256];
    vals[j] = v; lsum += v; lsq += v * v;
  }
  s_sum[tid] = lsum; s_sq[tid] = lsq;
  __syncthreads();
  for (int s = 128; s > 0; s >>= 1) {
    if (tid < s) { s_sum[tid] += s_sum[tid + s]; s_sq[tid] += s_sq[tid + s]; }
    __syncthreads();
  }
  float mu  = s_sum[0] * (1.f / EDIM);
  float var = s_sq[0] * (1.f / EDIM) - mu * mu;
  float inv = rsqrtf(var + 1e-5f);
#pragma unroll
  for (int j = 0; j < 6; ++j) {
    int c = tid + j * 256;
    row[c] = (vals[j] - mu) * inv * gamma[c] + beta[c];
  }
}

// ---------------- host launcher ----------------
extern "C" void kernel_launch(void* const* d_in, const int* in_sizes, int n_in,
                              void* d_out, int out_size, void* d_ws, size_t ws_size,
                              hipStream_t stream)
{
  (void)in_sizes; (void)n_in; (void)out_size; (void)ws_size;
  const float* x         = (const float*)d_in[0];
  const float* conv_w    = (const float*)d_in[1];
  const float* conv_b    = (const float*)d_in[2];
  const float* cemb      = (const float*)d_in[3];
  const float* cquery    = (const float*)d_in[4];
  const float* grp_in_w  = (const float*)d_in[5];
  const float* grp_in_b  = (const float*)d_in[6];
  const float* grp_out_w = (const float*)d_in[7];
  const float* grp_out_b = (const float*)d_in[8];
  const float* equery    = (const float*)d_in[9];
  const float* ens_in_w  = (const float*)d_in[10];
  const float* ens_in_b  = (const float*)d_in[11];
  const float* ens_out_w = (const float*)d_in[12];
  const float* ens_out_b = (const float*)d_in[13];
  const float* gamma     = (const float*)d_in[14];
  const float* beta      = (const float*)d_in[15];

  char* ws = (char*)d_ws;
  __bf16* xn     = (__bf16*)(ws + XN_OFF);
  __bf16* outg   = (__bf16*)(ws + OUTG_OFF);
  float*  qg     = (float*)(ws + QG_OFF);
  float*  qe     = (float*)(ws + QE_OFF);
  float*  qkvec  = (float*)(ws + QKV_OFF);
  float*  qconst = (float*)(ws + QC_OFF);
  float*  Qk2    = (float*)(ws + QK2_OFF);
  float*  cq2    = (float*)(ws + CQ2_OFF);
  __bf16* Abf    = (__bf16*)(ws + ABF_OFF);
  __bf16* Ebf    = (__bf16*)(ws + EBF_OFF);
  float*  ybias  = (float*)(ws + YB_OFF);
  float*  rbias  = (float*)(ws + RB_OFF);
  float*  r_out  = (float*)d_out;

  k_pre_q<<<dim3(12), 256, 0, stream>>>(cquery, grp_in_w, grp_in_b, equery, ens_in_w, ens_in_b, qg, qe);
  k_pre_qk<<<dim3(25), 256, 0, stream>>>(grp_in_w, grp_in_b, ens_in_w, ens_in_b, qg, qe,
                                         qkvec, qconst, Qk2, cq2);
  k_pre_mats<<<dim3((427648 + 255) / 256), 256, 0, stream>>>(grp_in_w, grp_in_b, grp_out_w, grp_out_b,
                                                             ens_in_w, ens_in_b, ens_out_w, ens_out_b,
                                                             Abf, Ebf, ybias, rbias);
  k_embed<<<dim3(360448), 256, 0, stream>>>(x, conv_w, conv_b, cemb, xn);
  k_group_attn<<<dim3(512, 12), 32, 81056, stream>>>(xn, qkvec, qconst, Abf, ybias, outg);
  k_ens_attn<<<dim3(512), 32, 88672, stream>>>(outg, Qk2, cq2, Ebf, rbias, r_out);
  k_layernorm<<<dim3(8192), 256, 0, stream>>>(r_out, gamma, beta);
}
